// LinearAttention_47614007443707
// MI455X (gfx1250) — matrix operand order repaired and checked by reading, not compiled, not run on hardware
//
#include <hip/hip_runtime.h>

// ---------------------------------------------------------------------------
// XCA attention forward for MI455X (gfx1250), bf16 WMMA pipeline with
// async global->LDS staging and double-buffered tiles.
// B=4, N=4096, DIM=1024, HEADS=16, DIM_HEAD=64.
// ---------------------------------------------------------------------------

typedef __attribute__((ext_vector_type(16))) __bf16 v16bf;
typedef __attribute__((ext_vector_type(8)))  __bf16 v8bf;
typedef __attribute__((ext_vector_type(4)))  __bf16 v4bf;
typedef __attribute__((ext_vector_type(8)))  float  v8f;
typedef __attribute__((ext_vector_type(4)))  int    v4i;

#define WMMA_BF16(a, b, c) \
    __builtin_amdgcn_wmma_f32_16x16x32_bf16(false, (a), false, (b), (short)0, (c), false, false)

#if __has_builtin(__builtin_amdgcn_global_load_async_to_lds_b128) && \
    __has_builtin(__builtin_amdgcn_s_wait_asynccnt)
#define HAVE_ASYNC_LDS 1
typedef __attribute__((address_space(1))) v4i gas_v4i;
typedef __attribute__((address_space(3))) v4i las_v4i;
#else
#define HAVE_ASYNC_LDS 0
#endif

// 16-byte global -> LDS copy; async (ASYNCcnt) when the toolchain has it.
__device__ inline void cp16(const __bf16* g, __bf16* l) {
#if HAVE_ASYNC_LDS
    __builtin_amdgcn_global_load_async_to_lds_b128(
        (gas_v4i*)g, (las_v4i*)l, 0, 0);
#else
    *(v8bf*)l = *(const v8bf*)g;
#endif
}

template <int N> __device__ inline void wait_async() {
#if HAVE_ASYNC_LDS
    __builtin_amdgcn_s_wait_asynccnt(N);
#endif
}

// ---- fragment loaders matching gfx1250 wave32 WMMA lane layouts -----------
// A (16x32 bf16): lanes 0-15 -> M=lane, elems 0..7 = K khalf+0..7,
// elems 8..15 = K khalf+16..23; khalf = 8 for lanes 16-31.
__device__ inline v16bf load_a_frag(const __bf16* base, int ldp) {
    int lane  = threadIdx.x & 31;
    int row   = lane & 15;
    int khalf = (lane >> 4) << 3;
    const __bf16* p = base + row * ldp + khalf;
    v8bf lo = *(const v8bf*)(p);
    v8bf hi = *(const v8bf*)(p + 16);
    v16bf a;
#pragma unroll
    for (int i = 0; i < 8; ++i) { a[i] = lo[i]; a[i + 8] = hi[i]; }
    return a;
}

// B (32x16 bf16): lanes 0-15 -> N=lane, elems 0..7 = K khalf+0..7,
// elems 8..15 = K khalf+16..23; khalf = 8 for lanes 16-31 (N=lane-16).
// Staged in LDS N-major (BT[n][k]) so the 16 K-values are contiguous.
__device__ inline v16bf load_b_frag_t(const __bf16* base, int ldp) {
    int lane  = threadIdx.x & 31;
    int col   = lane & 15;
    int kbase = (lane >> 4) << 3;
    const __bf16* p = base + col * ldp + kbase;
    v8bf lo = *(const v8bf*)(p);
    v8bf hi = *(const v8bf*)(p + 16);
    v16bf b;
#pragma unroll
    for (int i = 0; i < 8; ++i) { b[i] = lo[i]; b[i + 8] = hi[i]; }
    return b;
}

// C/D (16x16 f32, 8 VGPRs): VGPR r, lanes 0-15 -> M=r, N=lane;
// lanes 16-31 -> M=r+8, N=lane-16.
__device__ inline void store_c_frag(float* C, int ldc, v8f c) {
    int lane = threadIdx.x & 31;
    int col  = lane & 15;
    int rb   = (lane >> 4) << 3;
#pragma unroll
    for (int r = 0; r < 8; ++r) C[(size_t)(rb + r) * ldc + col] = c[r];
}

// ---------------------------------------------------------------------------
__global__ __launch_bounds__(256) void zero_f32_kernel(float* p, int n) {
    int i = blockIdx.x * 256 + threadIdx.x;
    if (i < n) p[i] = 0.f;
}

// f32 [K][N] -> bf16 [N][K] tiled transpose (one-time weight prep).
__global__ __launch_bounds__(256)
void cvt_transpose_kernel(const float* __restrict__ in, __bf16* __restrict__ out,
                          int K, int N) {
    __shared__ float tile[32][33];
    const int tx = threadIdx.x & 31, ty = threadIdx.x >> 5;
    const int n0 = blockIdx.x * 32, k0 = blockIdx.y * 32;
#pragma unroll
    for (int j = 0; j < 4; ++j)
        tile[ty + j * 8][tx] = in[(size_t)(k0 + ty + j * 8) * N + n0 + tx];
    __syncthreads();
#pragma unroll
    for (int j = 0; j < 4; ++j)
        out[(size_t)(n0 + ty + j * 8) * K + k0 + tx] = (__bf16)tile[tx][ty + j * 8];
}

// RMSNorm: xn = x / max(||x||,eps) * sqrt(1024) * gamma  -> bf16
__global__ __launch_bounds__(256) void rmsnorm_kernel(const float* __restrict__ x,
                                                      const float* __restrict__ g,
                                                      __bf16* __restrict__ xn) {
    __shared__ float red[8];
    const int row = blockIdx.x;
    const int t   = threadIdx.x;
    const float4 vx = *(const float4*)(x + (size_t)row * 1024 + t * 4);
    float ss = vx.x * vx.x + vx.y * vx.y + vx.z * vx.z + vx.w * vx.w;
#pragma unroll
    for (int m = 16; m >= 1; m >>= 1) ss += __shfl_xor(ss, m, 32);
    if ((t & 31) == 0) red[t >> 5] = ss;
    __syncthreads();
    if (t == 0) {
        float tot = 0.f;
#pragma unroll
        for (int i = 0; i < 8; ++i) tot += red[i];
        red[0] = 32.0f / fmaxf(sqrtf(tot), 1e-12f);   // sqrt(1024)=32
    }
    __syncthreads();
    const float s = red[0];
    const float4 gg = *(const float4*)(g + t * 4);
    v4bf o;
    o[0] = (__bf16)(vx.x * s * gg.x);
    o[1] = (__bf16)(vx.y * s * gg.y);
    o[2] = (__bf16)(vx.z * s * gg.z);
    o[3] = (__bf16)(vx.w * s * gg.w);
    *(v4bf*)(xn + (size_t)row * 1024 + t * 4) = o;
}

// ---------------------------------------------------------------------------
// Tiled bf16 WMMA GEMM: C[M,N] = A[M,K] @ Bt[N,K]^T.  128x128 tile, K-step 32,
// 8 waves, double-buffered async LDS staging, tail-peeled steady-state loop.
// MODE 0: plain f32 store.  MODE 1: QKV scatter (q,k,v -> (b,h,chan,n) packed)
// + per-channel sumsq atomics for q/k norms.
// ---------------------------------------------------------------------------
template <int MODE>
__global__ __launch_bounds__(256)
void gemm_bf16_kernel(const __bf16* __restrict__ A, const __bf16* __restrict__ Bt,
                      float* __restrict__ C, int M, int N, int K,
                      __bf16* __restrict__ qp, __bf16* __restrict__ kp,
                      __bf16* __restrict__ vp, float* __restrict__ ssq) {
    __shared__ __attribute__((aligned(16))) __bf16 As[2][128 * 40];
    __shared__ __attribute__((aligned(16))) __bf16 Bs[2][128 * 40];

    const int t = threadIdx.x, wave = t >> 5, lane = t & 31;
    const int m0 = blockIdx.y * 128, n0 = blockIdx.x * 128;
    const int wr = wave & 3, wc = wave >> 2;

    // per-thread staging chunks: c in {t, t+256}; row = c>>2, ko = (c&3)*8
    const int rA0 = t >> 2, ko0 = (t & 3) << 3;
    const int rA1 = (t + 256) >> 2, ko1 = ((t + 256) & 3) << 3;
    const __bf16* gA0 = A + (size_t)(m0 + rA0) * K + ko0;
    const __bf16* gA1 = A + (size_t)(m0 + rA1) * K + ko1;
    const __bf16* gB0 = Bt + (size_t)(n0 + rA0) * K + ko0;
    const __bf16* gB1 = Bt + (size_t)(n0 + rA1) * K + ko1;
    __bf16* lA0 = &As[0][rA0 * 40 + ko0];
    __bf16* lA1 = &As[0][rA1 * 40 + ko1];
    __bf16* lB0 = &Bs[0][rA0 * 40 + ko0];
    __bf16* lB1 = &Bs[0][rA1 * 40 + ko1];

    // issue one K-step of staging into buffer `buf`, advance global pointers
    auto stage = [&](int buf) {
        const int off = buf * (128 * 40);
        cp16(gA0, lA0 + off);
        cp16(gA1, lA1 + off);
        cp16(gB0, lB0 + off);
        cp16(gB1, lB1 + off);
        gA0 += 32; gA1 += 32; gB0 += 32; gB1 += 32;
    };

    v8f acc[2][4];
#pragma unroll
    for (int i = 0; i < 2; ++i)
#pragma unroll
        for (int j = 0; j < 4; ++j) acc[i][j] = (v8f){};

    auto compute = [&](int cur) {
        const __bf16* Ac = &As[cur][0];
        const __bf16* Bc = &Bs[cur][0];
        v16bf af0 = load_a_frag(Ac + (32 * wr) * 40, 40);
        v16bf af1 = load_a_frag(Ac + (32 * wr + 16) * 40, 40);
        v16bf bfr[4];
#pragma unroll
        for (int cf = 0; cf < 4; ++cf)
            bfr[cf] = load_b_frag_t(Bc + (64 * wc + 16 * cf) * 40, 40);
#pragma unroll
        for (int cf = 0; cf < 4; ++cf) {
            acc[0][cf] = WMMA_BF16(af0, bfr[cf], acc[0][cf]);
            acc[1][cf] = WMMA_BF16(af1, bfr[cf], acc[1][cf]);
        }
    };

    // prologue + steady state + peeled tail (no per-iteration conditionals)
    stage(0);
    int cur = 0;
    for (int k0 = 0; k0 < K - 32; k0 += 32) {
        stage(cur ^ 1);
        wait_async<4>();
        __syncthreads();
        compute(cur);
        __syncthreads();
        cur ^= 1;
    }
    wait_async<0>();
    __syncthreads();
    compute(cur);

    const int colb = n0 + 64 * wc;
    const int rowb = m0 + 32 * wr;

    if (MODE == 0) {
#pragma unroll
        for (int rf = 0; rf < 2; ++rf)
#pragma unroll
            for (int cf = 0; cf < 4; ++cf)
                store_c_frag(C + (size_t)(rowb + 16 * rf) * N + colb + 16 * cf, N,
                             acc[rf][cf]);
        return;
    }

    // MODE 1: q|k|v sections (all rows of a block share b; 1024 % 128 == 0).
    const int lane15 = lane & 15, rb = (lane >> 4) << 3;
    const int sec = colb >> 10;
    const int b   = m0 >> 12;
#pragma unroll
    for (int cf = 0; cf < 4; ++cf) {
        const int cc = (colb + 16 * cf + lane15) & 1023;
        const int h = cc >> 6, dd = cc & 63;
        const size_t bhd = ((size_t)(b * 16 + h) * 64 + dd);
        float sq = 0.f;
#pragma unroll
        for (int rf = 0; rf < 2; ++rf) {
            const int n = ((rowb + 16 * rf + rb) & 4095);
            v8bf pk;
#pragma unroll
            for (int r = 0; r < 8; ++r) {
                float vv = acc[rf][cf][r];
                sq += vv * vv;
                pk[r] = (__bf16)vv;
            }
            __bf16* dst = (sec == 0) ? qp : (sec == 1) ? kp : vp;
            *(v8bf*)(dst + bhd * 4096 + n) = pk;
        }
        if (sec == 0)      atomicAdd(&ssq[(b * 16 + h) * 64 + dd], sq);
        else if (sec == 1) atomicAdd(&ssq[4096 + (b * 16 + h) * 64 + dd], sq);
    }
}

// ---------------------------------------------------------------------------
// Per-(b,h) fused attention.  q,k,v all stored (b,h,chan,n) bf16.
// Phase 1: raw sim = q @ k^T over n (K=4096), async double-buffered tiles.
// Epilogue: apply factored-out l2 norms * exp(temperature) * 8, softmax.
// Phase 2: out = attn @ v streamed over n with register-pipelined LDS
// transpose of v; packed-bf16 transposed store into y (b,n,h*64+d).
// 128 threads = 4 waves; each wave owns one 16-row (d) strip.
// ---------------------------------------------------------------------------
__global__ __launch_bounds__(128)
void attn_kernel(const __bf16* __restrict__ q, const __bf16* __restrict__ k,
                 const __bf16* __restrict__ v, const float* __restrict__ ssq,
                 const float* __restrict__ temp, __bf16* __restrict__ y) {
    __shared__ __attribute__((aligned(16))) union {
        struct { __bf16 qA[2][64 * 40]; __bf16 kB[2][64 * 40]; } p1;
        float simS[64 * 68];
    } sh;
    __shared__ __attribute__((aligned(16))) __bf16 attnS[64 * 72];
    __shared__ __attribute__((aligned(16))) __bf16 vBt[2][64 * 72];
    __shared__ float qs[64], ks[64];

    const int bh = blockIdx.x;
    const int b = bh >> 4, h = bh & 15;
    const int t = threadIdx.x, wave = t >> 5, lane = t & 31;
    const size_t cbase = (size_t)bh * 64 * 4096;

    // per-thread staging chunks: c in {t, t+128}; row = c>>2, ko = (c&3)*8
    const int r0 = t >> 2, o0 = (t & 3) << 3;
    const int r1 = (t + 128) >> 2, o1 = ((t + 128) & 3) << 3;
    const __bf16* gq0 = q + cbase + (size_t)r0 * 4096 + o0;
    const __bf16* gq1 = q + cbase + (size_t)r1 * 4096 + o1;
    const __bf16* gk0 = k + cbase + (size_t)r0 * 4096 + o0;
    const __bf16* gk1 = k + cbase + (size_t)r1 * 4096 + o1;
    __bf16* lq0 = &sh.p1.qA[0][r0 * 40 + o0];
    __bf16* lq1 = &sh.p1.qA[0][r1 * 40 + o1];
    __bf16* lk0 = &sh.p1.kB[0][r0 * 40 + o0];
    __bf16* lk1 = &sh.p1.kB[0][r1 * 40 + o1];

    auto stage1 = [&](int buf) {
        const int off = buf * (64 * 40);
        cp16(gq0, lq0 + off);
        cp16(gq1, lq1 + off);
        cp16(gk0, lk0 + off);
        cp16(gk1, lk1 + off);
        gq0 += 32; gq1 += 32; gk0 += 32; gk1 += 32;
    };

    // ---- phase 1 ----
    v8f acc[4];
#pragma unroll
    for (int j = 0; j < 4; ++j) acc[j] = (v8f){};

    auto compute1 = [&](int cur) {
        v16bf af = load_a_frag(&sh.p1.qA[cur][0] + (16 * wave) * 40, 40);
        v16bf bfr[4];
#pragma unroll
        for (int cf = 0; cf < 4; ++cf)
            bfr[cf] = load_b_frag_t(&sh.p1.kB[cur][0] + (16 * cf) * 40, 40);
#pragma unroll
        for (int cf = 0; cf < 4; ++cf) acc[cf] = WMMA_BF16(af, bfr[cf], acc[cf]);
    };

    stage1(0);
    int cur = 0;
    for (int k0 = 0; k0 < 4096 - 32; k0 += 32) {
        stage1(cur ^ 1);
        wait_async<4>();
        __syncthreads();
        compute1(cur);
        __syncthreads();
        cur ^= 1;
    }
    wait_async<0>();
    __syncthreads();
    compute1(cur);
    __syncthreads();

    {   // spill sim to LDS (f32); tile buffers are dead now (union overlay)
        int col = lane & 15, rb = (lane >> 4) << 3;
#pragma unroll
        for (int cf = 0; cf < 4; ++cf)
#pragma unroll
            for (int r = 0; r < 8; ++r)
                sh.simS[(16 * wave + rb + r) * 68 + 16 * cf + col] = acc[cf][r];
    }
    if (t < 64) {
        qs[t] = __expf(temp[h]) / fmaxf(sqrtf(ssq[bh * 64 + t]), 1e-12f);
        ks[t] = 1.f / fmaxf(sqrtf(ssq[4096 + bh * 64 + t]), 1e-12f);
    }
    __syncthreads();

    if (t < 64) {   // softmax over e, one thread per row d
        const int d = t;
        const float qsc = qs[d] * 8.0f;                 // ATTN_SCALE
        float mx = -3.4e38f;
        for (int e = 0; e < 64; ++e)
            mx = fmaxf(mx, sh.simS[d * 68 + e] * qsc * ks[e]);
        float sum = 0.f;
        for (int e = 0; e < 64; ++e)
            sum += __expf(sh.simS[d * 68 + e] * qsc * ks[e] - mx);
        const float inv = 1.f / sum;
        for (int e = 0; e < 64; ++e)
            attnS[d * 72 + e] =
                (__bf16)(__expf(sh.simS[d * 68 + e] * qsc * ks[e] - mx) * inv);
    }
    __syncthreads();

    // ---- phase 2 ----
    v16bf aA[2];
    aA[0] = load_a_frag(attnS + (16 * wave) * 72, 72);        // K(e) = 0..31
    aA[1] = load_a_frag(attnS + (16 * wave) * 72 + 32, 72);   // K(e) = 32..63

    // per-thread v chunks: c = t + j*128 -> er = c>>3, nb = (c&7)*8
    const __bf16* gv[4];
#pragma unroll
    for (int j = 0; j < 4; ++j) {
        const int c = t + j * 128;
        gv[j] = v + cbase + (size_t)(c >> 3) * 4096 + ((c & 7) << 3);
    }
    v8bf vr[4];
    auto loadV = [&]() {
#pragma unroll
        for (int j = 0; j < 4; ++j) { vr[j] = *(const v8bf*)gv[j]; gv[j] += 64; }
    };
    auto scatterV = [&](int buf) {
#pragma unroll
        for (int j = 0; j < 4; ++j) {
            const int c = t + j * 128;
            const int er = c >> 3, nb = (c & 7) << 3;
#pragma unroll
            for (int jj = 0; jj < 8; ++jj) vBt[buf][(nb + jj) * 72 + er] = vr[j][jj];
        }
    };

    const int col = lane & 15, rb = (lane >> 4) << 3;
    const int dbase = 16 * wave + rb;
    auto compute2 = [&](int c2, int n0) {
        v8f o[4];
#pragma unroll
        for (int j = 0; j < 4; ++j) o[j] = (v8f){};
        v16bf bfr[4];
#pragma unroll
        for (int ks2 = 0; ks2 < 2; ++ks2) {
#pragma unroll
            for (int cf = 0; cf < 4; ++cf)
                bfr[cf] = load_b_frag_t(&vBt[c2][0] + (16 * cf) * 72 + ks2 * 32, 72);
#pragma unroll
            for (int cf = 0; cf < 4; ++cf) o[cf] = WMMA_BF16(aA[ks2], bfr[cf], o[cf]);
        }
#pragma unroll
        for (int cf = 0; cf < 4; ++cf) {    // packed transposed store
            const int n = n0 + 16 * cf + col;
            v8bf pk;
#pragma unroll
            for (int r = 0; r < 8; ++r) pk[r] = (__bf16)o[cf][r];
            *(v8bf*)(y + ((size_t)(b * 4096 + n)) * 1024 + h * 64 + dbase) = pk;
        }
    };

    loadV();
    scatterV(0);
    __syncthreads();
    int c2 = 0;
    for (int n0 = 0; n0 < 4096 - 64; n0 += 64) {
        loadV();                             // global loads overlap the WMMAs
        compute2(c2, n0);
        scatterV(c2 ^ 1);
        __syncthreads();
        c2 ^= 1;
    }
    compute2(c2, 4096 - 64);
}

// ---------------------------------------------------------------------------
extern "C" void kernel_launch(void* const* d_in, const int* in_sizes, int n_in,
                              void* d_out, int out_size, void* d_ws, size_t ws_size,
                              hipStream_t stream) {
    (void)in_sizes; (void)n_in; (void)out_size; (void)ws_size;
    const float* x     = (const float*)d_in[0];
    const float* gamma = (const float*)d_in[1];
    const float* w_qkv = (const float*)d_in[2];
    const float* temp  = (const float*)d_in[3];
    const float* w_out = (const float*)d_in[4];
    float* out = (float*)d_out;

    char* ws = (char*)d_ws;
    __bf16* xn  = (__bf16*)ws;  ws += (size_t)16384 * 1024 * 2;
    __bf16* wqT = (__bf16*)ws;  ws += (size_t)3072 * 1024 * 2;   // [N][K]
    __bf16* woT = (__bf16*)ws;  ws += (size_t)1024 * 1024 * 2;   // [N][K]
    __bf16* qb  = (__bf16*)ws;  ws += (size_t)4 * 16 * 64 * 4096 * 2;
    __bf16* kb  = (__bf16*)ws;  ws += (size_t)4 * 16 * 64 * 4096 * 2;
    __bf16* vb  = (__bf16*)ws;  ws += (size_t)4 * 16 * 64 * 4096 * 2;
    __bf16* y   = (__bf16*)ws;  ws += (size_t)16384 * 1024 * 2;
    float*  ssq = (float*)ws;   // 8192 f32: [0..4095]=q sumsq, [4096..8191]=k

    zero_f32_kernel<<<32, 256, 0, stream>>>(ssq, 8192);
    cvt_transpose_kernel<<<dim3(96, 32), 256, 0, stream>>>(w_qkv, wqT, 1024, 3072);
    cvt_transpose_kernel<<<dim3(32, 32), 256, 0, stream>>>(w_out, woT, 1024, 1024);
    rmsnorm_kernel<<<16384, 256, 0, stream>>>(x, gamma, xn);
    gemm_bf16_kernel<1><<<dim3(24, 128), 256, 0, stream>>>(
        xn, wqT, nullptr, 16384, 3072, 1024, qb, kb, vb, ssq);
    attn_kernel<<<64, 128, 0, stream>>>(qb, kb, vb, ssq, temp, y);
    gemm_bf16_kernel<0><<<dim3(8, 128), 256, 0, stream>>>(
        y, woT, out, 16384, 1024, 1024, nullptr, nullptr, nullptr, nullptr);
}
